// LinearAttentionTransformerLM_14224931684796
// MI455X (gfx1250) — compile-verified, matrix operand order checked by reading, not run on hardware
//
#include <hip/hip_runtime.h>
#include <hip/hip_bf16.h>
#include <math.h>
#include <stdint.h>

// ---- model dims (fixed by reference) ----
#define SEQ   1024
#define DIM   512
#define NH    8
#define DHEAD 64
#define FFD   2048
#define NL    4
#define VOCAB 32000

typedef __attribute__((ext_vector_type(16))) _Float16 v16h;
typedef __attribute__((ext_vector_type(8)))  _Float16 v8h;
typedef __attribute__((ext_vector_type(8)))  float    v8f;
typedef __attribute__((__vector_size__(16))) int      v4i;   // matches builtin param type

// ---- CDNA5 async global->LDS path (ASYNCcnt) --------------------------------
#if defined(__has_builtin)
#  if __has_builtin(__builtin_amdgcn_global_load_async_to_lds_b128)
#    define USE_ASYNC_LDS 1
#  endif
#endif
#ifndef USE_ASYNC_LDS
#  define USE_ASYNC_LDS 0
#endif

// Rebuild addrspace-qualified int4 pointers from flat addresses (LDS flat addr
// low 32 bits == LDS byte offset on CDNA5, ISA 10.2 aperture mapping; flat ==
// global address for AS1).
#define AS1P(p) ((__attribute__((address_space(1))) v4i*)(unsigned long long)(uintptr_t)(p))
#define AS3P(p) ((__attribute__((address_space(3))) v4i*)(unsigned)(uintptr_t)(p))

__device__ __forceinline__ void wait_asynccnt0() {
#if defined(__has_builtin)
#  if __has_builtin(__builtin_amdgcn_s_wait_asynccnt)
  __builtin_amdgcn_s_wait_asynccnt(0);
#  else
  asm volatile("s_wait_asynccnt 0x0" ::: "memory");
#  endif
#else
  asm volatile("s_wait_asynccnt 0x0" ::: "memory");
#endif
}

// ---------------------------------------------------------------------------
// Embedding: x[n,d] = tok_emb[tokens[n],d] + pos_emb[n,d]
// ---------------------------------------------------------------------------
__global__ void embed_kernel(const int* __restrict__ tokens,
                             const float* __restrict__ tok_emb,
                             const float* __restrict__ pos_emb,
                             float* __restrict__ x) {
  int idx = blockIdx.x * 256 + threadIdx.x;
  int n = idx >> 9;
  int d = idx & (DIM - 1);
  x[idx] = tok_emb[(size_t)tokens[n] * DIM + d] + pos_emb[idx];
}

// ---------------------------------------------------------------------------
// Weight transpose + fp32 -> f16 convert:  in[K][N] -> out[N][K]
// ---------------------------------------------------------------------------
__global__ void wt_transpose_f16(const float* __restrict__ in,
                                 _Float16* __restrict__ out, int K, int N) {
  __shared__ _Float16 tile[32][33];
  int nb = blockIdx.x * 32, kb = blockIdx.y * 32;
  int tx = threadIdx.x, ty = threadIdx.y;             // 32 x 8
#pragma unroll
  for (int i = 0; i < 32; i += 8)
    tile[ty + i][tx] = (_Float16)in[(size_t)(kb + ty + i) * N + nb + tx];
  __syncthreads();
#pragma unroll
  for (int i = 0; i < 32; i += 8)
    out[(size_t)(nb + ty + i) * K + kb + tx] = tile[tx][ty + i];
}

// ---------------------------------------------------------------------------
// LayerNorm (per row of [SEQ, DIM]) -> f16 activations for GEMM A-operand
// ---------------------------------------------------------------------------
__global__ void layernorm_f16_kernel(const float* __restrict__ x,
                                     const float* __restrict__ g,
                                     const float* __restrict__ b,
                                     _Float16* __restrict__ out) {
  __shared__ float r1[128], r2[128];
  const int row = blockIdx.x, t = threadIdx.x;
  const float* xr = x + (size_t)row * DIM;
  float s = 0.f, s2 = 0.f;
  for (int d = t; d < DIM; d += 128) { float v = xr[d]; s += v; s2 += v * v; }
  r1[t] = s; r2[t] = s2; __syncthreads();
#pragma unroll
  for (int st = 64; st > 0; st >>= 1) {
    if (t < st) { r1[t] += r1[t + st]; r2[t] += r2[t + st]; }
    __syncthreads();
  }
  const float mean = r1[0] * (1.f / DIM);
  const float var  = r2[0] * (1.f / DIM) - mean * mean;
  const float rstd = rsqrtf(var + 1e-5f);
  for (int d = t; d < DIM; d += 128)
    out[(size_t)row * DIM + d] = (_Float16)((xr[d] - mean) * rstd * g[d] + b[d]);
}

// ---------------------------------------------------------------------------
// WMMA GEMM:  C[M,N] (op)= A[M,K](f16) * Bt[N,K](f16)^T + bias
//   EPI 0: C(f32) = val          EPI 1: C(f32) += val          EPI 2: Ch(f16) = gelu(val)
// Block: 256 threads = 8 wave32 (4x2). Block tile 128x128, wave tile 32x64
// (2 A-frags x 4 B-frags = 8 WMMA / K-step). Double-buffered LDS; async
// global->LDS staging (ASYNCcnt) overlapped with WMMA when available.
// M,N multiples of 128; K multiple of 32.
// ---------------------------------------------------------------------------
template <int EPI>
__global__ void __launch_bounds__(256)
gemm_f16_wmma(const _Float16* __restrict__ A, const _Float16* __restrict__ Bt,
              const float* __restrict__ bias, float* __restrict__ C,
              _Float16* __restrict__ Ch, int M, int N, int K) {
  __shared__ __align__(16) _Float16 lA[2][128][40];   // 32 K-halves + pad
  __shared__ __align__(16) _Float16 lB[2][128][40];

  const int tid  = threadIdx.x;
  const int lane = tid & 31;
  const int wid  = tid >> 5;
  const int wr   = wid >> 1;                 // 0..3  (32-row slab)
  const int wc   = wid & 1;                  // 0..1  (64-col slab)
  const int rowBase = blockIdx.y * 128;
  const int colBase = blockIdx.x * 128;

  // staging: 128 rows x 32 halves per operand; thread -> 32B of one row
  const int srow = tid >> 1;                 // 0..127
  const int skq  = (tid & 1) * 16;           // 0 or 16 halves
  const _Float16* gA = A  + (size_t)(rowBase + srow) * K + skq;
  const _Float16* gB = Bt + (size_t)(colBase + srow) * K + skq;

  // fragment coords (16-bit 16x32 layout: lane%16 = row/col, lane/16 = K-half)
  const int mrow = lane & 15;
  const int kb   = (lane >> 4) * 8;

  v8f acc[2][4];
#pragma unroll
  for (int a = 0; a < 2; ++a)
#pragma unroll
    for (int j = 0; j < 4; ++j) acc[a][j] = (v8f){};

  // ---- prologue: stage tile 0 into buffer 0 ----
#if USE_ASYNC_LDS
  __builtin_amdgcn_global_load_async_to_lds_b128(AS1P(gA),     AS3P(&lA[0][srow][skq]),     0, 0);
  __builtin_amdgcn_global_load_async_to_lds_b128(AS1P(gA + 8), AS3P(&lA[0][srow][skq + 8]), 0, 0);
  __builtin_amdgcn_global_load_async_to_lds_b128(AS1P(gB),     AS3P(&lB[0][srow][skq]),     0, 0);
  __builtin_amdgcn_global_load_async_to_lds_b128(AS1P(gB + 8), AS3P(&lB[0][srow][skq + 8]), 0, 0);
  wait_asynccnt0();
#else
  *(v8h*)&lA[0][srow][skq]     = *(const v8h*)(gA);
  *(v8h*)&lA[0][srow][skq + 8] = *(const v8h*)(gA + 8);
  *(v8h*)&lB[0][srow][skq]     = *(const v8h*)(gB);
  *(v8h*)&lB[0][srow][skq + 8] = *(const v8h*)(gB + 8);
#endif
  __syncthreads();

  int buf = 0;
  for (int k0 = 0; k0 < K; k0 += 32) {
    const bool more = (k0 + 32) < K;
#if USE_ASYNC_LDS
    if (more) {   // issue next tile while this one computes
      __builtin_amdgcn_global_load_async_to_lds_b128(AS1P(gA + k0 + 32), AS3P(&lA[buf ^ 1][srow][skq]),     0, 0);
      __builtin_amdgcn_global_load_async_to_lds_b128(AS1P(gA + k0 + 40), AS3P(&lA[buf ^ 1][srow][skq + 8]), 0, 0);
      __builtin_amdgcn_global_load_async_to_lds_b128(AS1P(gB + k0 + 32), AS3P(&lB[buf ^ 1][srow][skq]),     0, 0);
      __builtin_amdgcn_global_load_async_to_lds_b128(AS1P(gB + k0 + 40), AS3P(&lB[buf ^ 1][srow][skq + 8]), 0, 0);
    }
#else
    v8h ra0, ra1, rb0, rb1;
    if (more) {
      ra0 = *(const v8h*)(gA + k0 + 32);
      ra1 = *(const v8h*)(gA + k0 + 40);
      rb0 = *(const v8h*)(gB + k0 + 32);
      rb1 = *(const v8h*)(gB + k0 + 40);
    }
#endif

    union Frag { v16h v; v8h h[2]; } af[2], bf[4];
#pragma unroll
    for (int a = 0; a < 2; ++a) {
      const int r = wr * 32 + a * 16 + mrow;
      af[a].h[0] = *(const v8h*)&lA[buf][r][kb];
      af[a].h[1] = *(const v8h*)&lA[buf][r][kb + 16];
    }
#pragma unroll
    for (int j = 0; j < 4; ++j) {
      const int r = wc * 64 + j * 16 + mrow;
      bf[j].h[0] = *(const v8h*)&lB[buf][r][kb];
      bf[j].h[1] = *(const v8h*)&lB[buf][r][kb + 16];
    }
#pragma unroll
    for (int a = 0; a < 2; ++a)
#pragma unroll
      for (int j = 0; j < 4; ++j)
        acc[a][j] = __builtin_amdgcn_wmma_f32_16x16x32_f16(false, af[a].v, false, bf[j].v,
                                                           (short)0, acc[a][j], false, false);

#if USE_ASYNC_LDS
    if (more) wait_asynccnt0();
#else
    if (more) {
      *(v8h*)&lA[buf ^ 1][srow][skq]     = ra0;
      *(v8h*)&lA[buf ^ 1][srow][skq + 8] = ra1;
      *(v8h*)&lB[buf ^ 1][srow][skq]     = rb0;
      *(v8h*)&lB[buf ^ 1][srow][skq + 8] = rb1;
    }
#endif
    __syncthreads();
    buf ^= 1;
  }

  // C/D layout: VGPR i -> M = i + 8*(lane/16), N = lane%16
  const int row0 = rowBase + wr * 32 + (lane >> 4) * 8;
  const int colW = colBase + wc * 64 + (lane & 15);
#pragma unroll
  for (int a = 0; a < 2; ++a) {
#pragma unroll
    for (int j = 0; j < 4; ++j) {
      const int col = colW + j * 16;
      const float bb = bias ? bias[col] : 0.f;
#pragma unroll
      for (int i = 0; i < 8; ++i) {
        const size_t idx = (size_t)(row0 + a * 16 + i) * N + col;
        const float v = acc[a][j][i] + bb;
        if (EPI == 0)      C[idx] = v;
        else if (EPI == 1) C[idx] += v;
        else Ch[idx] = (_Float16)(0.5f * v * (1.f + erff(v * 0.70710678118f)));
      }
    }
  }
}

// ---------------------------------------------------------------------------
// Causal linear attention over fused QKV buffer [SEQ][3*DIM].
// One block per head, 64 threads; thread t owns ctx[:,t] in 64 VGPRs.
// ---------------------------------------------------------------------------
__global__ void __launch_bounds__(64)
linear_attn_kernel(const float* __restrict__ QKV, _Float16* __restrict__ Aout) {
  __shared__ float qs[DHEAD], ks[DHEAD], red[DHEAD];
  const int h = blockIdx.x, t = threadIdx.x;
  const int co = h * DHEAD + t;
  float ctx[DHEAD];
#pragma unroll
  for (int d = 0; d < DHEAD; ++d) ctx[d] = 0.f;
  float kcum = 0.f;

  for (int n = 0; n < SEQ; ++n) {
    const size_t o = (size_t)n * (3 * DIM) + co;
    const float qv = QKV[o];
    const float kv = QKV[o + DIM];
    const float vv = QKV[o + 2 * DIM];

    red[t] = qv; __syncthreads();
#pragma unroll
    for (int s = 32; s > 0; s >>= 1) {
      if (t < s) red[t] = fmaxf(red[t], red[t + s]);
      __syncthreads();
    }
    const float qmax = red[0]; __syncthreads();
    const float qe = __expf(qv - qmax);
    red[t] = qe; __syncthreads();
#pragma unroll
    for (int s = 32; s > 0; s >>= 1) {
      if (t < s) red[t] += red[t + s];
      __syncthreads();
    }
    const float qsum = red[0]; __syncthreads();
    const float qsm = qe / qsum * 0.125f;            // * DHEAD^-0.5
    const float ke  = __expf(kv);
    kcum += ke;

    qs[t] = qsm; ks[t] = ke;
    red[t] = qsm * kcum; __syncthreads();            // denom = sum_d q*k_cum
#pragma unroll
    for (int s = 32; s > 0; s >>= 1) {
      if (t < s) red[t] += red[t + s];
      __syncthreads();
    }
    const float dinv = 1.f / fmaxf(red[0], 1e-3f);

    float acc = 0.f;
#pragma unroll
    for (int d = 0; d < DHEAD; ++d) {                // inclusive scan
      ctx[d] = fmaf(ks[d], vv, ctx[d]);
      acc    = fmaf(qs[d], ctx[d], acc);
    }
    Aout[(size_t)n * DIM + co] = (_Float16)(acc * dinv);
    __syncthreads();
  }
}

// ---------------------------------------------------------------------------
// Host orchestration
// ---------------------------------------------------------------------------
extern "C" void kernel_launch(void* const* d_in, const int* in_sizes, int n_in,
                              void* d_out, int out_size, void* d_ws, size_t ws_size,
                              hipStream_t stream) {
  (void)in_sizes; (void)n_in; (void)out_size; (void)ws_size;
  const int*   tokens  = (const int*)  d_in[0];
  const float* tok_emb = (const float*)d_in[1];
  const float* pos_emb = (const float*)d_in[2];
  const float* Wq  = (const float*)d_in[3];
  const float* Wk  = (const float*)d_in[4];
  const float* Wv  = (const float*)d_in[5];
  const float* Wo  = (const float*)d_in[6];
  const float* bo  = (const float*)d_in[7];
  const float* ln1g = (const float*)d_in[8];
  const float* ln1b = (const float*)d_in[9];
  const float* W1  = (const float*)d_in[10];
  const float* b1  = (const float*)d_in[11];
  const float* W2  = (const float*)d_in[12];
  const float* b2  = (const float*)d_in[13];
  const float* ln2g = (const float*)d_in[14];
  const float* ln2b = (const float*)d_in[15];
  const float* lnfg = (const float*)d_in[16];
  const float* lnfb = (const float*)d_in[17];
  const float* Wout = (const float*)d_in[18];
  const float* bout = (const float*)d_in[19];
  float* out = (float*)d_out;

  char* p = (char*)d_ws;
  auto carve = [&](size_t bytes) -> char* {
    char* r = p; p += (bytes + 255) & ~(size_t)255; return r;
  };
  _Float16* QKVT  = (_Float16*)carve((size_t)NL * 3 * DIM * DIM * 2);  // fused [1536][512] per layer
  _Float16* WoT   = (_Float16*)carve((size_t)NL * DIM * DIM * 2);
  _Float16* W1T   = (_Float16*)carve((size_t)NL * DIM * FFD * 2);
  _Float16* W2T   = (_Float16*)carve((size_t)NL * DIM * FFD * 2);
  _Float16* WoutT = (_Float16*)carve((size_t)DIM * VOCAB * 2);
  float*    x     = (float*)   carve((size_t)SEQ * DIM * 4);
  _Float16* h16   = (_Float16*)carve((size_t)SEQ * DIM * 2);
  float*    qkvf  = (float*)   carve((size_t)SEQ * 3 * DIM * 4);
  _Float16* a16   = (_Float16*)carve((size_t)SEQ * DIM * 2);
  _Float16* ff16  = (_Float16*)carve((size_t)SEQ * FFD * 2);

  const dim3 tb(32, 8);
  for (int i = 0; i < NL; ++i) {
    const size_t wo = (size_t)i * DIM * DIM;
    const size_t fo = (size_t)i * DIM * FFD;
    _Float16* qkvt = QKVT + (size_t)i * 3 * DIM * DIM;
    wt_transpose_f16<<<dim3(DIM / 32, DIM / 32), tb, 0, stream>>>(Wq + wo, qkvt,                   DIM, DIM);
    wt_transpose_f16<<<dim3(DIM / 32, DIM / 32), tb, 0, stream>>>(Wk + wo, qkvt + DIM * DIM,       DIM, DIM);
    wt_transpose_f16<<<dim3(DIM / 32, DIM / 32), tb, 0, stream>>>(Wv + wo, qkvt + 2 * DIM * DIM,   DIM, DIM);
    wt_transpose_f16<<<dim3(DIM / 32, DIM / 32), tb, 0, stream>>>(Wo + wo, WoT + wo,               DIM, DIM);
    wt_transpose_f16<<<dim3(FFD / 32, DIM / 32), tb, 0, stream>>>(W1 + fo, W1T + fo,               DIM, FFD);
    wt_transpose_f16<<<dim3(DIM / 32, FFD / 32), tb, 0, stream>>>(W2 + fo, W2T + fo,               FFD, DIM);
  }
  wt_transpose_f16<<<dim3(VOCAB / 32, DIM / 32), tb, 0, stream>>>(Wout, WoutT, DIM, VOCAB);

  embed_kernel<<<SEQ * DIM / 256, 256, 0, stream>>>(tokens, tok_emb, pos_emb, x);

  const dim3 gD(DIM / 128, SEQ / 128);      // (4, 8)
  for (int i = 0; i < NL; ++i) {
    const size_t wo = (size_t)i * DIM * DIM;
    const size_t fo = (size_t)i * DIM * FFD;
    layernorm_f16_kernel<<<SEQ, 128, 0, stream>>>(x, ln1g + i * DIM, ln1b + i * DIM, h16);
    gemm_f16_wmma<0><<<dim3(3 * DIM / 128, SEQ / 128), 256, 0, stream>>>(
        h16, QKVT + (size_t)i * 3 * DIM * DIM, nullptr, qkvf, nullptr, SEQ, 3 * DIM, DIM);
    linear_attn_kernel<<<NH, 64, 0, stream>>>(qkvf, a16);
    gemm_f16_wmma<1><<<gD, 256, 0, stream>>>(a16, WoT + wo, bo + i * DIM, x, nullptr, SEQ, DIM, DIM);
    layernorm_f16_kernel<<<SEQ, 128, 0, stream>>>(x, ln2g + i * DIM, ln2b + i * DIM, h16);
    gemm_f16_wmma<2><<<dim3(FFD / 128, SEQ / 128), 256, 0, stream>>>(
        h16, W1T + fo, b1 + i * FFD, nullptr, ff16, SEQ, FFD, DIM);
    gemm_f16_wmma<1><<<gD, 256, 0, stream>>>(ff16, W2T + fo, b2 + i * DIM, x, nullptr, SEQ, DIM, FFD);
  }
  layernorm_f16_kernel<<<SEQ, 128, 0, stream>>>(x, lnfg, lnfb, h16);
  gemm_f16_wmma<0><<<dim3(VOCAB / 128, SEQ / 128), 256, 0, stream>>>(
      h16, WoutT, bout, out, nullptr, SEQ, VOCAB, DIM);
}
